// ScaledPrototypeLayer_27513560498836
// MI455X (gfx1250) — matrix operand order repaired and checked by reading, not compiled, not run on hardware
//
#include <hip/hip_runtime.h>
#include <hip/hip_bf16.h>
#include <stdint.h>

// Problem constants (from reference setup_inputs)
#define NN    16384
#define DD    768
#define PP    2048
#define CC    1000
#define CPAD  1024

typedef __attribute__((ext_vector_type(16))) __bf16 v16bf;
typedef __attribute__((ext_vector_type(8)))  float  v8f;

union FragU { uint4 q[2]; v16bf v; };

__device__ __forceinline__ uint16_t f2bf(float f) {
    // round-to-nearest-even f32 -> bf16
    uint32_t u = __float_as_uint(f);
    uint32_t r = u + 0x7FFFu + ((u >> 16) & 1u);
    return (uint16_t)(r >> 16);
}

// ---- A fragment: 16x32 bf16, row-major source, ld = row stride (elems) ----
// lane<16 : row = lane,    K = {0..7} in v[0..7], {16..23} in v[8..15]
// lane>=16: row = lane-16, K = {8..15},{24..31}
__device__ __forceinline__ v16bf load_A(const uint16_t* __restrict__ base, int ld,
                                        int row0, int k0, int lane) {
    const uint16_t* p = base + (size_t)(row0 + (lane & 15)) * ld + k0 + ((lane >> 4) << 3);
    FragU u;
    u.q[0] = *(const uint4*)(p);
    u.q[1] = *(const uint4*)(p + 16);
    return u.v;
}

// ---- B fragment: 32x16 bf16. Column n of B = row n of a row-major [n,K] array ----
// lane<16 : col = lane,    K = {0..7} in v[0..7], {16..23} in v[8..15]
// lane>=16: col = lane-16, K = {8..15},{24..31}
__device__ __forceinline__ v16bf load_B(const uint16_t* __restrict__ base, int ld,
                                        int col0, int k0, int lane) {
    const uint16_t* p = base + (size_t)(col0 + (lane & 15)) * ld + k0 + ((lane >> 4) << 3);
    FragU u;
    u.q[0] = *(const uint4*)(p);
    u.q[1] = *(const uint4*)(p + 16);
    return u.v;
}

// ---------------- Kernel 1/2: L2-normalize rows, emit bf16 ----------------
__global__ void spl_normalize_rows(const float* __restrict__ in,
                                   uint16_t* __restrict__ outbf,
                                   int rows, int cols) {
    int row  = blockIdx.x * (blockDim.x >> 5) + (threadIdx.x >> 5);
    int lane = threadIdx.x & 31;
    if (row >= rows) return;
    const float* r = in + (size_t)row * cols;
    float s = 0.0f;
    for (int j = lane; j < cols; j += 32) { float x = r[j]; s += x * x; }
    #pragma unroll
    for (int off = 16; off > 0; off >>= 1) s += __shfl_xor(s, off, 32);
    float inv = 1.0f / fmaxf(sqrtf(s), 1e-12f);
    uint16_t* o = outbf + (size_t)row * cols;
    for (int j = lane; j < cols; j += 32) o[j] = f2bf(r[j] * inv);
}

// ------- Kernel 3: transpose prototype_class [K=2048, C=1000] f32 ->
//                   pct [CPAD=1024, 2048] bf16 (zero-pad c>=1000) -------
__global__ void spl_transpose_pc(const float* __restrict__ pc,
                                 uint16_t* __restrict__ pct) {
    size_t idx = (size_t)blockIdx.x * blockDim.x + threadIdx.x;
    if (idx >= (size_t)CPAD * PP) return;
    int k = (int)(idx & (PP - 1));
    int c = (int)(idx >> 11);          // PP = 2048 = 1<<11
    float v = (c < CC) ? pc[(size_t)k * CC + c] : 0.0f;
    pct[idx] = f2bf(v);
}

// ------- Kernel 4: GEMM1 (h_norm @ p_norm^T) fused distance/exp epilogue -------
// grid = (PP/256, NN/128), block = 256 (8 waves, 2Mx4N). Wave tile: 64(M) x 64(N).
__global__ void __launch_bounds__(256, 1)
spl_gemm1_act(const uint16_t* __restrict__ hbf,
              const uint16_t* __restrict__ pbf,
              const float* __restrict__ temperature,
              uint16_t* __restrict__ act) {
    const int lane = threadIdx.x & 31;
    const int wid  = threadIdx.x >> 5;
    const int rowBase = blockIdx.y * 128 + (wid >> 2) * 64;  // into N (=16384)
    const int colBase = blockIdx.x * 256 + (wid & 3) * 64;   // into P (=2048)

    v8f acc[4][4] = {};
    #pragma unroll 1
    for (int k0 = 0; k0 < DD; k0 += 32) {
        if (k0 + 32 < DD) {
            __builtin_prefetch(hbf + (size_t)(rowBase + (lane & 15)) * DD + k0 + 32, 0, 1);
            __builtin_prefetch(pbf + (size_t)(colBase + (lane & 15)) * DD + k0 + 32, 0, 1);
        }
        v16bf a[4], b[4];
        #pragma unroll
        for (int mi = 0; mi < 4; ++mi) a[mi] = load_A(hbf, DD, rowBase + mi * 16, k0, lane);
        #pragma unroll
        for (int ni = 0; ni < 4; ++ni) b[ni] = load_B(pbf, DD, colBase + ni * 16, k0, lane);
        #pragma unroll
        for (int mi = 0; mi < 4; ++mi)
            #pragma unroll
            for (int ni = 0; ni < 4; ++ni)
                acc[mi][ni] = __builtin_amdgcn_wmma_f32_16x16x32_bf16(
                    false, a[mi], false, b[ni], (short)0, acc[mi][ni], false, false);
    }

    // tau = softplus(temperature)
    float t   = temperature[0];
    float tau = (t > 15.0f) ? t : log1pf(__expf(t));

    const int lhalf = lane >> 4;   // D layout: M = r + 8*lhalf, Ncol = lane&15
    const int lmod  = lane & 15;
    #pragma unroll
    for (int mi = 0; mi < 4; ++mi)
        #pragma unroll
        for (int ni = 0; ni < 4; ++ni)
            #pragma unroll
            for (int r = 0; r < 8; ++r) {
                int n = rowBase + mi * 16 + r + lhalf * 8;
                int p = colBase + ni * 16 + lmod;
                float dot  = acc[mi][ni][r];
                float d2   = fmaxf(2.0f - 2.0f * dot, 0.0f);  // ||a||=||b||=1
                float dist = sqrtf(d2);
                float a    = __expf(-tau * dist);
                act[(size_t)n * PP + p] = f2bf(a);
            }
}

// ------- Kernel 5: GEMM2 (act @ prototype_class) -> f32 logits -------
// grid = (CPAD/256, NN/128), block = 256. Wave tile: 64(M) x 64(C).
__global__ void __launch_bounds__(256, 1)
spl_gemm2(const uint16_t* __restrict__ act,
          const uint16_t* __restrict__ pct,
          float* __restrict__ out) {
    const int lane = threadIdx.x & 31;
    const int wid  = threadIdx.x >> 5;
    const int rowBase = blockIdx.y * 128 + (wid >> 2) * 64;  // into N
    const int colBase = blockIdx.x * 256 + (wid & 3) * 64;   // into CPAD

    v8f acc[4][4] = {};
    #pragma unroll 1
    for (int k0 = 0; k0 < PP; k0 += 32) {
        if (k0 + 32 < PP) {
            __builtin_prefetch(act + (size_t)(rowBase + (lane & 15)) * PP + k0 + 32, 0, 1);
            __builtin_prefetch(pct + (size_t)(colBase + (lane & 15)) * PP + k0 + 32, 0, 1);
        }
        v16bf a[4], b[4];
        #pragma unroll
        for (int mi = 0; mi < 4; ++mi) a[mi] = load_A(act, PP, rowBase + mi * 16, k0, lane);
        #pragma unroll
        for (int ni = 0; ni < 4; ++ni) b[ni] = load_B(pct, PP, colBase + ni * 16, k0, lane);
        #pragma unroll
        for (int mi = 0; mi < 4; ++mi)
            #pragma unroll
            for (int ni = 0; ni < 4; ++ni)
                acc[mi][ni] = __builtin_amdgcn_wmma_f32_16x16x32_bf16(
                    false, a[mi], false, b[ni], (short)0, acc[mi][ni], false, false);
    }

    const int lhalf = lane >> 4;
    const int lmod  = lane & 15;
    #pragma unroll
    for (int mi = 0; mi < 4; ++mi)
        #pragma unroll
        for (int ni = 0; ni < 4; ++ni) {
            int c = colBase + ni * 16 + lmod;
            if (c >= CC) continue;  // padded columns
            #pragma unroll
            for (int r = 0; r < 8; ++r) {
                int n = rowBase + mi * 16 + r + lhalf * 8;
                out[(size_t)n * CC + c] = acc[mi][ni][r];
            }
        }
}

extern "C" void kernel_launch(void* const* d_in, const int* in_sizes, int n_in,
                              void* d_out, int out_size, void* d_ws, size_t ws_size,
                              hipStream_t stream) {
    (void)in_sizes; (void)n_in; (void)out_size; (void)ws_size;
    const float* h    = (const float*)d_in[0];   // [16384, 768]
    const float* prot = (const float*)d_in[1];   // [2048, 768]
    const float* pc   = (const float*)d_in[2];   // [2048, 1000]
    const float* temp = (const float*)d_in[3];   // scalar
    float* out = (float*)d_out;                  // [16384, 1000]

    // Workspace layout (bf16 = uint16_t), ~100 MB total:
    uint16_t* ws   = (uint16_t*)d_ws;
    uint16_t* h_bf = ws;                                  // N*D   = 12,582,912
    uint16_t* p_bf = h_bf + (size_t)NN * DD;              // P*D   =  1,572,864
    uint16_t* pct  = p_bf + (size_t)PP * DD;              // CPAD*P=  2,097,152
    uint16_t* actb = pct  + (size_t)CPAD * PP;            // N*P   = 33,554,432

    // 1) normalize h -> bf16  (one wave per row, 8 waves per block)
    spl_normalize_rows<<<NN / 8, 256, 0, stream>>>(h, h_bf, NN, DD);
    // 2) normalize prototypes -> bf16
    spl_normalize_rows<<<PP / 8, 256, 0, stream>>>(prot, p_bf, PP, DD);
    // 3) transpose + pad prototype_class -> bf16 [CPAD, P]
    spl_transpose_pc<<<(CPAD * PP) / 256, 256, 0, stream>>>(pc, pct);
    // 4) GEMM1 + fused distance/exp epilogue -> act bf16 [N, P]
    spl_gemm1_act<<<dim3(PP / 256, NN / 128), 256, 0, stream>>>(h_bf, p_bf, temp, actb);
    // 5) GEMM2 -> f32 logits [N, C]
    spl_gemm2<<<dim3(CPAD / 256, NN / 128), 256, 0, stream>>>(actb, pct, out);
}
